// ProgramSliceGNN_15857019257220
// MI455X (gfx1250) — matrix operand order repaired and checked by reading, not compiled, not run on hardware
//
#include <hip/hip_runtime.h>
#include <hip/hip_bf16.h>

#define NN 50000     // nodes
#define NE 300000    // edges
#define NR 3         // relations

typedef __attribute__((ext_vector_type(16))) __bf16 v16bf;
typedef __attribute__((ext_vector_type(8)))  float  v8f;

union Frag {
  v16bf v;
  unsigned short s[16];
  uint4 u4[2];
};

__device__ __forceinline__ unsigned short f2bf(float f) {
  unsigned int u = __float_as_uint(f);
  u = (u + 0x7fffu + ((u >> 16) & 1u)) >> 16;   // round-to-nearest-even
  return (unsigned short)u;
}
__device__ __forceinline__ float bf2f(unsigned short h) {
  return __uint_as_float(((unsigned int)h) << 16);
}

// f32 [R,C] -> bf16 [Rp,Cp], zero-padded, optional ReLU (activations; row-major kept).
__global__ void cvt_bf16_kernel(const float* __restrict__ src, unsigned short* __restrict__ dst,
                                int R, int C, int Rp, int Cp, int relu) {
  long long idx = (long long)blockIdx.x * blockDim.x + threadIdx.x;
  long long tot = (long long)Rp * Cp;
  if (idx >= tot) return;
  int r = (int)(idx / Cp), c = (int)(idx % Cp);
  float v = (r < R && c < C) ? src[(long long)r * C + c] : 0.0f;
  if (relu) v = fmaxf(v, 0.0f);
  dst[idx] = f2bf(v);
}

// f32 B[K,N] -> bf16 BT[N,Kp] (transposed, K zero-padded to Kp).
__global__ void cvt_bf16_t_kernel(const float* __restrict__ src, unsigned short* __restrict__ dst,
                                  int K, int N, int Kp) {
  long long idx = (long long)blockIdx.x * blockDim.x + threadIdx.x;
  long long tot = (long long)N * Kp;
  if (idx >= tot) return;
  int n = (int)(idx / Kp), k = (int)(idx % Kp);
  float v = (k < K) ? src[(long long)k * N + n] : 0.0f;
  dst[idx] = f2bf(v);
}

// C[M,N] = A[M,K](bf16,row-major,lda) x B (given as BT[N,Kp],bf16,row-major ldbt)
// (+bias)(+relu), f32 out. M%16==0, N%64==0, K%32==0 (zero-padded).
// Block = 256 threads = 8 waves; block tile 256x64; wave tile 32x64 (8 WMMA accumulators).
__global__ __launch_bounds__(256)
void gemm_bf16_wmma(const unsigned short* __restrict__ A, int lda,
                    const unsigned short* __restrict__ BT, int ldbt,
                    float* __restrict__ C, int ldc,
                    const float* __restrict__ bias,
                    int M, int N, int K, int relu) {
  // BT tile: 64 output-columns x 32 K-halves, stored row-major [n][k] -> fragments contiguous
  __shared__ __align__(16) unsigned short Bs[64 * 32];

  const int tid  = threadIdx.x;
  const int wave = tid >> 5;
  const int lane = tid & 31;
  const int mm   = lane & 15;
  const int hi   = lane >> 4;

  const int m_base = blockIdx.x * 256 + wave * 32;
  const int n0     = blockIdx.y * 64;
  // per-16-row clamps (M%16==0) so OOB waves still run barriers with valid addresses
  const int mb0 = m_base;
  const int mb1 = m_base + 16;
  const int ml0 = (mb0 + 16 <= M) ? mb0 : (M - 16);
  const int ml1 = (mb1 + 16 <= M) ? mb1 : (M - 16);

  v8f acc[2][4] = {};

  // cooperative BT tile load: 2048 halves, 8 contiguous per thread
  const int ld_idx = tid * 8;
  const int ld_row = ld_idx >> 5;    // 0..63  (output column within tile)
  const int ld_k   = ld_idx & 31;    // 0,8,16,24

  const unsigned short* a0p = &A[(long long)(ml0 + mm) * lda + hi * 8];
  const unsigned short* a1p = &A[(long long)(ml1 + mm) * lda + hi * 8];
  const unsigned short* btp = &BT[(long long)(n0 + ld_row) * ldbt + ld_k];

  for (int k0 = 0; k0 < K; k0 += 32) {
    __syncthreads();
    *reinterpret_cast<uint4*>(&Bs[ld_idx]) = *reinterpret_cast<const uint4*>(btp + k0);
    if (k0 + 32 < K) __builtin_prefetch(btp + k0 + 32, 0, 1);   // global_prefetch_b8
    __syncthreads();

    // A fragments (ISA 7.12.2, 16-bit A 16x32): half t -> K = t + (t&8) + hi*8
    Frag a0, a1;
    a0.u4[0] = *reinterpret_cast<const uint4*>(a0p + k0);
    a0.u4[1] = *reinterpret_cast<const uint4*>(a0p + k0 + 16);
    a1.u4[0] = *reinterpret_cast<const uint4*>(a1p + k0);
    a1.u4[1] = *reinterpret_cast<const uint4*>(a1p + k0 + 16);

#pragma unroll
    for (int j = 0; j < 4; ++j) {
      // B fragment (32x16): lane column = j*16+mm, half t -> K = t + (t&8) + hi*8
      Frag b;
      const unsigned short* bs = &Bs[(j * 16 + mm) * 32 + hi * 8];
      b.u4[0] = *reinterpret_cast<const uint4*>(bs);
      b.u4[1] = *reinterpret_cast<const uint4*>(bs + 16);
#ifdef __HIP_DEVICE_COMPILE__
      acc[0][j] = __builtin_amdgcn_wmma_f32_16x16x32_bf16(false, a0.v, false, b.v,
                                                          (short)0, acc[0][j], false, false);
      acc[1][j] = __builtin_amdgcn_wmma_f32_16x16x32_bf16(false, a1.v, false, b.v,
                                                          (short)0, acc[1][j], false, false);
#endif
    }
  }

#pragma unroll
  for (int g = 0; g < 2; ++g) {
    int mb = m_base + g * 16;
    if (mb + 16 > M) continue;   // sub-tile lies past M (computed on clamped rows)
#pragma unroll
    for (int j = 0; j < 4; ++j) {
      int col = n0 + j * 16 + mm;
      float bv = bias ? bias[col] : 0.0f;
#pragma unroll
      for (int r = 0; r < 8; ++r) {     // D layout: VGPR r -> row r + hi*8, col = lane&15
        int row = mb + hi * 8 + r;
        float v = acc[g][j][r] + bv;
        if (relu) v = fmaxf(v, 0.0f);
        C[(long long)row * ldc + col] = v;
      }
    }
  }
}

__global__ void edge_count_kernel(const int* __restrict__ dst, const int* __restrict__ et,
                                  float* __restrict__ cnt) {
  int e = blockIdx.x * blockDim.x + threadIdx.x;
  if (e >= NE) return;
  atomicAdd(&cnt[et[e] * NN + dst[e]], 1.0f);
}

__global__ void inv_kernel(float* __restrict__ cnt) {
  int i = blockIdx.x * blockDim.x + threadIdx.x;
  if (i >= NR * NN) return;
  cnt[i] = 1.0f / fmaxf(cnt[i], 1.0f);
}

// out[dst] += msg[src] * inv[r,dst] for edges of relation r. One block per edge.
__global__ void scatter_kernel(const float* __restrict__ msg,
                               const int* __restrict__ src, const int* __restrict__ dst,
                               const int* __restrict__ et, const float* __restrict__ inv,
                               float* __restrict__ out, int D, int r) {
  int e = blockIdx.x;
  if (et[e] != r) return;
  int s = src[e], d = dst[e];
  float sc = inv[r * NN + d];
  const float* mrow = &msg[(long long)s * D];
  float* orow = &out[(long long)d * D];
  for (int c = threadIdx.x; c < D; c += blockDim.x)
    atomicAdd(&orow[c], mrow[c] * sc);
}

// final 256 -> 3 projection, f32 output
__global__ void head_out_kernel(const unsigned short* __restrict__ xbf,
                                const float* __restrict__ w, const float* __restrict__ b,
                                float* __restrict__ out) {
  int i = blockIdx.x * blockDim.x + threadIdx.x;
  if (i >= NN) return;
  float a0 = b[0], a1 = b[1], a2 = b[2];
  const unsigned short* xr = &xbf[(long long)i * 256];
  for (int c = 0; c < 256; ++c) {
    float xv = bf2f(xr[c]);
    a0 += xv * w[c * 3 + 0];
    a1 += xv * w[c * 3 + 1];
    a2 += xv * w[c * 3 + 2];
  }
  out[i * 3 + 0] = a0; out[i * 3 + 1] = a1; out[i * 3 + 2] = a2;
}

extern "C" void kernel_launch(void* const* d_in, const int* in_sizes, int n_in,
                              void* d_out, int out_size, void* d_ws, size_t ws_size,
                              hipStream_t stream) {
  (void)in_sizes; (void)n_in; (void)out_size; (void)ws_size;

  const float* x     = (const float*)d_in[0];
  const int*   ei    = (const int*)d_in[1];
  const int*   etype = (const int*)d_in[2];
  const float* w_in  = (const float*)d_in[3];
  const float* b_in  = (const float*)d_in[4];
  const float* w1    = (const float*)d_in[5];
  const float* root1 = (const float*)d_in[6];
  const float* b1    = (const float*)d_in[7];
  const float* w2    = (const float*)d_in[8];
  const float* root2 = (const float*)d_in[9];
  const float* b2    = (const float*)d_in[10];
  const float* w3    = (const float*)d_in[11];
  const float* root3 = (const float*)d_in[12];
  const float* b3    = (const float*)d_in[13];
  const float* w_o1  = (const float*)d_in[14];
  const float* b_o1  = (const float*)d_in[15];
  const float* w_o2  = (const float*)d_in[16];
  const float* b_o2  = (const float*)d_in[17];
  const float* w_o3  = (const float*)d_in[18];
  const float* b_o3  = (const float*)d_in[19];
  const int* esrc = ei;
  const int* edst = ei + NE;

  // ---- workspace carve-out -------------------------------------------------
  char* ws = (char*)d_ws;
  size_t off = 0;
  auto take = [&](size_t bytes) -> char* {
    char* p = ws + off;
    off += (bytes + 255) & ~(size_t)255;
    return p;
  };
  unsigned short* XBF = (unsigned short*)take((size_t)NN * 1024 * 2);  // bf16 activations
  float* OUT = (float*)take((size_t)NN * 1024 * 4);                    // f32 layer output
  float* MSG = (float*)take((size_t)NN * 1024 * 4);                    // f32 relation messages (reused)
  float* INV = (float*)take((size_t)NR * NN * 4);                      // per-(rel,dst) 1/max(cnt,1)
  // all weights stored TRANSPOSED bf16: BT[N, Kp]
  unsigned short* WIN = (unsigned short*)take((size_t)1024 * 800 * 2);
  unsigned short* R1  = (unsigned short*)take((size_t)1024 * 1024 * 2);
  unsigned short* W1  = (unsigned short*)take((size_t)3 * 1024 * 1024 * 2);
  unsigned short* R2  = (unsigned short*)take((size_t)512 * 1024 * 2);
  unsigned short* W2  = (unsigned short*)take((size_t)3 * 512 * 1024 * 2);
  unsigned short* R3  = (unsigned short*)take((size_t)256 * 512 * 2);
  unsigned short* W3  = (unsigned short*)take((size_t)3 * 256 * 512 * 2);
  unsigned short* WO1 = (unsigned short*)take((size_t)512 * 256 * 2);
  unsigned short* WO2 = (unsigned short*)take((size_t)256 * 512 * 2);

  auto cvt = [&](const float* s, unsigned short* d, int R, int C, int Rp, int Cp, int relu) {
    long long tot = (long long)Rp * Cp;
    hipLaunchKernelGGL(cvt_bf16_kernel, dim3((unsigned)((tot + 255) / 256)), dim3(256), 0, stream,
                       s, d, R, C, Rp, Cp, relu);
  };
  auto cvtT = [&](const float* s, unsigned short* d, int K, int N, int Kp) {
    long long tot = (long long)N * Kp;
    hipLaunchKernelGGL(cvt_bf16_t_kernel, dim3((unsigned)((tot + 255) / 256)), dim3(256), 0, stream,
                       s, d, K, N, Kp);
  };
  auto gemm = [&](const unsigned short* A, int lda, const unsigned short* BT, int ldbt,
                  float* Cc, int ldc, const float* bias, int M, int N, int K, int relu) {
    hipLaunchKernelGGL(gemm_bf16_wmma, dim3((M + 255) / 256, N / 64), dim3(256), 0, stream,
                       A, lda, BT, ldbt, Cc, ldc, bias, M, N, K, relu);
  };

  // ---- edge degree counts (layer-independent), inverted once ---------------
  hipMemsetAsync(INV, 0, (size_t)NR * NN * 4, stream);
  hipLaunchKernelGGL(edge_count_kernel, dim3((NE + 255) / 256), dim3(256), 0, stream, edst, etype, INV);
  hipLaunchKernelGGL(inv_kernel, dim3((NR * NN + 255) / 256), dim3(256), 0, stream, INV);

  // ---- bf16 transposed weight conversion (K padded to multiple of 32) ------
  cvtT(w_in,  WIN, 782, 1024, 800);
  cvtT(root1, R1, 1024, 1024, 1024);
  for (int r = 0; r < NR; ++r)
    cvtT(w1 + (size_t)r * 1024 * 1024, W1 + (size_t)r * 1024 * 1024, 1024, 1024, 1024);
  cvtT(root2, R2, 1024, 512, 1024);
  for (int r = 0; r < NR; ++r)
    cvtT(w2 + (size_t)r * 1024 * 512, W2 + (size_t)r * 512 * 1024, 1024, 512, 1024);
  cvtT(root3, R3, 512, 256, 512);
  for (int r = 0; r < NR; ++r)
    cvtT(w3 + (size_t)r * 512 * 256, W3 + (size_t)r * 256 * 512, 512, 256, 512);
  cvtT(w_o1, WO1, 256, 512, 256);
  cvtT(w_o2, WO2, 512, 256, 512);

  // ---- input projection: relu(x @ w_in + b_in) -----------------------------
  cvt(x, XBF, NN, 782, NN, 800, 0);                          // pad K 782 -> 800
  gemm(XBF, 800, WIN, 800, OUT, 1024, b_in, NN, 1024, 800, 1);
  cvt(OUT, XBF, NN, 1024, NN, 1024, 0);

  // ---- RGCN layer ----------------------------------------------------------
  auto rgcn = [&](int Din, int Dout, const unsigned short* Wbf, const unsigned short* Rbf,
                  const float* bias) {
    gemm(XBF, Din, Rbf, Din, OUT, Dout, bias, NN, Dout, Din, 0);       // root + bias
    for (int r = 0; r < NR; ++r) {
      gemm(XBF, Din, Wbf + (size_t)r * Dout * Din, Din, MSG, Dout, nullptr, NN, Dout, Din, 0);
      hipLaunchKernelGGL(scatter_kernel, dim3(NE), dim3(256), 0, stream,
                         MSG, esrc, edst, etype, INV, OUT, Dout, r);   // mean-scaled scatter-add
    }
    cvt(OUT, XBF, NN, Dout, NN, Dout, 1);                              // ReLU + bf16 for next layer
  };

  rgcn(1024, 1024, W1, R1, b1);
  rgcn(1024,  512, W2, R2, b2);
  rgcn( 512,  256, W3, R3, b3);

  // ---- MLP head ------------------------------------------------------------
  gemm(XBF, 256, WO1, 256, OUT, 512, b_o1, NN, 512, 256, 1);
  cvt(OUT, XBF, NN, 512, NN, 512, 0);
  gemm(XBF, 512, WO2, 512, OUT, 256, b_o2, NN, 256, 512, 1);
  cvt(OUT, XBF, NN, 256, NN, 256, 0);
  hipLaunchKernelGGL(head_out_kernel, dim3((NN + 255) / 256), dim3(256), 0, stream,
                     XBF, w_o3, b_o3, (float*)d_out);
}
